// Model_39676907885106
// MI455X (gfx1250) — matrix operand order repaired and checked by reading, not compiled, not run on hardware
//
#include <hip/hip_runtime.h>
#include <hip/hip_bf16.h>

// ---------------------------------------------------------------------------
// Types for CDNA5 WMMA (wave32): v_wmma_f32_16x16x32_bf16
// ---------------------------------------------------------------------------
typedef __attribute__((ext_vector_type(16))) __bf16 v16bf;
typedef __attribute__((ext_vector_type(8)))  float  v8f;
typedef __attribute__((ext_vector_type(2)))  __bf16 bfx2;

// Optional CDNA5 async global->LDS copies (ASYNCcnt path). Guarded so the
// kernel still compiles (with the plain copy path) if the toolchain lacks
// the builtins. The builtin's params are AS1/AS3 pointers to a 16-byte int
// vector (exact type taken from the compiler diagnostic).
#if __has_builtin(__builtin_amdgcn_global_load_async_to_lds_b128) && \
    __has_builtin(__builtin_amdgcn_s_wait_asynccnt)
#define USE_ASYNC_LDS 1
#else
#define USE_ASYNC_LDS 0
#endif

#if USE_ASYNC_LDS
typedef int v4i __attribute__((vector_size(16)));
typedef __attribute__((address_space(1))) v4i* as1_v4i_ptr;
typedef __attribute__((address_space(3))) v4i* as3_v4i_ptr;

__device__ inline void async_copy_b128(const void* g, void* l) {
    __builtin_amdgcn_global_load_async_to_lds_b128(
        (as1_v4i_ptr)(v4i*)(void*)g, (as3_v4i_ptr)(v4i*)l, 0, 0);
}
#endif

__device__ inline v8f v8f_zero() {
    v8f z;
#pragma unroll
    for (int i = 0; i < 8; ++i) z[i] = 0.0f;
    return z;
}

__device__ inline v8f wmma_bf16(v16bf a, v16bf b, v8f c) {
    // (neg_a, A, neg_b, B, c_mod, C, reuse_a, reuse_b)
    return __builtin_amdgcn_wmma_f32_16x16x32_bf16(false, a, false, b,
                                                   (short)0, c, false, false);
}

// A-matrix fragment: 16x32 bf16, row-major source with row stride `stride`.
// ISA layout (16-bit A 16x32): lanes 0-15 row m=lane, VGPR j holds
// K = 2*(j&3) + 16*(j>>2) (+8 for lanes 16-31). Pairs are K-contiguous ->
// 32-bit LDS reads the compiler fuses into ds_load_b128.
__device__ inline v16bf load_a_frag(const __bf16* base, int stride, int lane) {
    const int m = lane & 15, g = lane >> 4;
    const __bf16* row = base + m * stride;
    v16bf a;
#pragma unroll
    for (int j = 0; j < 8; ++j) {
        const int k = 2 * (j & 3) + 16 * (j >> 2) + 8 * g;
        bfx2 p = *(const bfx2*)(row + k);
        a[2 * j] = p.x;
        a[2 * j + 1] = p.y;
    }
    return a;
}

// B-matrix fragment: 32x16 bf16 (KxN), sourced from B^T stored [N][K]
// row-major with row stride `stride`. Lane l: column n = l&15; VGPR j holds
// K = 2*(j&3) + 16*(j>>2) (+8 for lanes 16-31), matching the A fragment.
__device__ inline v16bf load_b_frag(const __bf16* base, int stride, int lane) {
    const int n = lane & 15, g = lane >> 4;
    const __bf16* row = base + n * stride;
    v16bf b;
#pragma unroll
    for (int j = 0; j < 8; ++j) {
        bfx2 p = *(const bfx2*)(row + 2 * (j & 3) + 16 * (j >> 2) + 8 * g);
        b[2 * j] = p.x;
        b[2 * j + 1] = p.y;
    }
    return b;
}

// C/D 16x16 f32 layout: lane l holds column n = l&15; VGPR j is row
// m = j + 8*(l>>4).

// ---------------------------------------------------------------------------
// Kernel 1 & 3: C[M][N] = A[M][K] * Bt[N][K]^T  (A, Bt fp32 in global,
// converted to bf16 on the LDS fill). 256 threads = 8 waves (4x2 wave grid),
// BM=128 BN=128 BK=32; wave (wr,wc) owns a 32x64 stripe = 2x4 WMMA tiles ->
// 8 WMMA per k-step. LDS tiles are double-buffered: the next K-slice is
// register-prefetched before the WMMA burst, one barrier per k-step.
// mode 0: scatter into bf16 Q (pre-scaled) / K / V workspace (N = 3*1024)
// mode 1: fp32 out + bias (N = 1024)
// ---------------------------------------------------------------------------
__global__ __launch_bounds__(256) void gemm_bf16_kernel(
    const float* __restrict__ A, const float* __restrict__ Bt,
    int K, int N, int mode, float qscale,
    __bf16* __restrict__ q_out, __bf16* __restrict__ k_out,
    __bf16* __restrict__ v_out,
    float* __restrict__ c_out, const float* __restrict__ bias) {
    __shared__ __bf16 sA[2][128][40];  // 128x32 tiles, stride 40 vs conflicts
    __shared__ __bf16 sB[2][128][40];

    const int tid = threadIdx.x, lane = tid & 31, w = tid >> 5;
    const int wr = w >> 1, wc = w & 1;  // 4x2 wave grid
    const long long m0 = (long long)blockIdx.y * 128;
    const long long n0 = (long long)blockIdx.x * 128;

    const int r_ld = tid >> 3, c_ld = tid & 7;  // this thread's slot in a tile

    v8f acc0[4], acc1[4];
#pragma unroll
    for (int t = 0; t < 4; ++t) { acc0[t] = v8f_zero(); acc1[t] = v8f_zero(); }

    float4 ra[4], rb[4];
    const int steps = K / 32;

    // prologue: stage k-slice 0 through registers into LDS buffer 0
#pragma unroll
    for (int p = 0; p < 4; ++p) {
        const int r = r_ld + p * 32;
        ra[p] = *(const float4*)(A  + (m0 + r) * K + 4 * c_ld);
        rb[p] = *(const float4*)(Bt + (n0 + r) * K + 4 * c_ld);
    }
#pragma unroll
    for (int p = 0; p < 4; ++p) {
        const int r = r_ld + p * 32;
        bfx2 a0; a0.x = (__bf16)ra[p].x; a0.y = (__bf16)ra[p].y;
        bfx2 a1; a1.x = (__bf16)ra[p].z; a1.y = (__bf16)ra[p].w;
        *(bfx2*)&sA[0][r][4 * c_ld]     = a0;
        *(bfx2*)&sA[0][r][4 * c_ld + 2] = a1;
        bfx2 b0; b0.x = (__bf16)rb[p].x; b0.y = (__bf16)rb[p].y;
        bfx2 b1; b1.x = (__bf16)rb[p].z; b1.y = (__bf16)rb[p].w;
        *(bfx2*)&sB[0][r][4 * c_ld]     = b0;
        *(bfx2*)&sB[0][r][4 * c_ld + 2] = b1;
    }
    __syncthreads();

    int cur = 0;
    for (int kt = 0; kt < steps; ++kt) {
        const bool more = (kt + 1 < steps);
        if (more) {  // issue next-slice global loads; latency hides under WMMA
            const int k0 = (kt + 1) * 32;
#pragma unroll
            for (int p = 0; p < 4; ++p) {
                const int r = r_ld + p * 32;
                ra[p] = *(const float4*)(A  + (m0 + r) * K + k0 + 4 * c_ld);
                rb[p] = *(const float4*)(Bt + (n0 + r) * K + k0 + 4 * c_ld);
            }
        }

        v16bf a0 = load_a_frag(&sA[cur][wr * 32][0], 40, lane);
        v16bf a1 = load_a_frag(&sA[cur][wr * 32 + 16][0], 40, lane);
#pragma unroll
        for (int t = 0; t < 4; ++t) {
            v16bf b = load_b_frag(&sB[cur][wc * 64 + t * 16][0], 40, lane);
            acc0[t] = wmma_bf16(a0, b, acc0[t]);
            acc1[t] = wmma_bf16(a1, b, acc1[t]);
        }

        if (more) {
            const int nxt = cur ^ 1;
#pragma unroll
            for (int p = 0; p < 4; ++p) {
                const int r = r_ld + p * 32;
                bfx2 a0p; a0p.x = (__bf16)ra[p].x; a0p.y = (__bf16)ra[p].y;
                bfx2 a1p; a1p.x = (__bf16)ra[p].z; a1p.y = (__bf16)ra[p].w;
                *(bfx2*)&sA[nxt][r][4 * c_ld]     = a0p;
                *(bfx2*)&sA[nxt][r][4 * c_ld + 2] = a1p;
                bfx2 b0p; b0p.x = (__bf16)rb[p].x; b0p.y = (__bf16)rb[p].y;
                bfx2 b1p; b1p.x = (__bf16)rb[p].z; b1p.y = (__bf16)rb[p].w;
                *(bfx2*)&sB[nxt][r][4 * c_ld]     = b0p;
                *(bfx2*)&sB[nxt][r][4 * c_ld + 2] = b1p;
            }
            __syncthreads();
            cur = nxt;
        }
    }

    // Epilogue
    const int nl = lane & 15, gh = lane >> 4;
#pragma unroll
    for (int mi = 0; mi < 2; ++mi) {
#pragma unroll
        for (int t = 0; t < 4; ++t) {
            const long long e = n0 + wc * 64 + t * 16 + nl;
#pragma unroll
            for (int jj = 0; jj < 8; ++jj) {
                const long long r = m0 + wr * 32 + mi * 16 + jj + 8 * gh;
                const float val = mi ? acc1[t][jj] : acc0[t][jj];
                if (mode == 0) {
                    const int seg = (int)(e >> 10);
                    const long long col = e & 1023;
                    if (seg == 0)      q_out[r * 1024 + col] = (__bf16)(val * qscale);
                    else if (seg == 1) k_out[r * 1024 + col] = (__bf16)val;
                    else               v_out[r * 1024 + col] = (__bf16)val;
                } else {
                    c_out[r * (long long)N + e] = val + bias[e];
                }
            }
        }
    }
}

// ---------------------------------------------------------------------------
// Kernel 2: flash attention, single head, head dim 1024, Q pre-scaled.
// One block = 16 query rows, 256 threads = 8 waves. Key blocks of 32.
// LDS (dynamic, ~185 KB — CDNA5 WGP has 320 KB):
//   sQ  [16][1040] bf16, sK [32][1040] bf16, sVt [1024][34] bf16 (transposed)
//   sSp [8][16*32] f32 per-wave score partials (deterministic reduction)
//   sS  [16*32] f32 scores, sCorr/sL [16] f32, sP [16][32] bf16
// Score phase: wave w contracts D-slice [128w,128w+128) (2 tiles x 4 k-steps).
// PV phase: wave w owns output columns [128w,128w+128) (8 tiles, 1 k-step).
// Q/K tile fills use GLOBAL_LOAD_ASYNC_TO_LDS_B128 when available.
// ---------------------------------------------------------------------------
#define QS 1040
#define KS 1040
#define VS 34
#define FA_SMEM_BYTES 189056
#define LOG2E 1.4426950408889634f

extern __shared__ char fa_smem[];

__global__ __launch_bounds__(256) void flash_attn_kernel(
    const __bf16* __restrict__ Q, const __bf16* __restrict__ Km,
    const __bf16* __restrict__ Vm, float* __restrict__ O, int S) {
    __bf16* sQ  = (__bf16*)fa_smem;
    __bf16* sK  = sQ + 16 * QS;               // elem 16640
    __bf16* sVt = sK + 32 * KS;               // elem 49920
    float*  sSp = (float*)(sVt + 1024 * VS);  // byte 169472
    float*  sS    = sSp + 8 * 512;
    float*  sCorr = sS + 512;
    float*  sL    = sCorr + 16;
    __bf16* sP  = (__bf16*)(sL + 16);

    const int tid = threadIdx.x, lane = tid & 31, w = tid >> 5;
    const long long qrow0 = (long long)blockIdx.y * S + (long long)blockIdx.x * 16;

    // Load Q block [16][1024] bf16
#if USE_ASYNC_LDS
    {
        const char* qgc = (const char*)(Q + qrow0 * 1024);
        for (int i = tid; i < 16 * 128; i += 256) {     // 16B chunks
            const int r = i >> 7, c = i & 127;
            async_copy_b128(qgc + r * 2048 + c * 16,
                            (char*)sQ + r * (QS * 2) + c * 16);
        }
    }
#else
    {
        const unsigned int* qg = (const unsigned int*)(Q + qrow0 * 1024);
        for (int i = tid; i < 16 * 512; i += 256) {
            const int r = i >> 9, c = i & 511;
            *(unsigned int*)&sQ[r * QS + 2 * c] = qg[r * 512 + c];
        }
    }
#endif

    v8f acc[8];
#pragma unroll
    for (int t = 0; t < 8; ++t) acc[t] = v8f_zero();
    float m_run = -3.0e38f, l_run = 0.0f;

    for (int j0 = 0; j0 < S; j0 += 32) {
        __syncthreads();  // prior PV reads of sVt/sP done; Q visible on iter 0
        const long long kvbase = ((long long)blockIdx.y * S + j0) * 1024;
#if USE_ASYNC_LDS
        {
            const char* kgc = (const char*)(Km + kvbase);
            for (int i = tid; i < 32 * 128; i += 256) {  // 16B chunks
                const int r = i >> 7, c = i & 127;
                async_copy_b128(kgc + r * 2048 + c * 16,
                                (char*)sK + r * (KS * 2) + c * 16);
            }
        }
#else
        {
            const unsigned int* kg = (const unsigned int*)(Km + kvbase);
            for (int i = tid; i < 32 * 512; i += 256) {
                const int r = i >> 9, c = i & 511;
                *(unsigned int*)&sK[r * KS + 2 * c] = kg[r * 512 + c];
            }
        }
#endif
        {   // V transposed into sVt (manual: async copies cannot transpose)
            const unsigned int* vg = (const unsigned int*)(Vm + kvbase);
            for (int i = tid; i < 32 * 512; i += 256) {
                const int r = i >> 9, c = i & 511;
                const unsigned int pv = vg[r * 512 + c];
                sVt[(2 * c) * VS + r]     = __builtin_bit_cast(__bf16, (unsigned short)(pv & 0xffffu));
                sVt[(2 * c + 1) * VS + r] = __builtin_bit_cast(__bf16, (unsigned short)(pv >> 16));
            }
        }
#if USE_ASYNC_LDS
        __builtin_amdgcn_s_wait_asynccnt(0);
#endif
        __syncthreads();

        // --- scores: partial over this wave's 128-wide D slice ---
        v8f sc0 = v8f_zero(), sc1 = v8f_zero();
#pragma unroll
        for (int kk = 0; kk < 4; ++kk) {
            const int kb = w * 128 + kk * 32;
            v16bf a  = load_a_frag(sQ + kb, QS, lane);
            v16bf b0 = load_b_frag(sK + kb, KS, lane);
            sc0 = wmma_bf16(a, b0, sc0);
            v16bf b1 = load_b_frag(sK + 16 * KS + kb, KS, lane);
            sc1 = wmma_bf16(a, b1, sc1);
        }
        {
            const int n = lane & 15, gh = lane >> 4;
            float* slice = sSp + w * 512;
#pragma unroll
            for (int jj = 0; jj < 8; ++jj) {
                const int mrow = jj + 8 * gh;
                slice[mrow * 32 + n]      = sc0[jj];
                slice[mrow * 32 + 16 + n] = sc1[jj];
            }
        }
        __syncthreads();
        // deterministic cross-wave reduction
        for (int i = tid; i < 512; i += 256) {
            float s = 0.0f;
#pragma unroll
            for (int ww = 0; ww < 8; ++ww) s += sSp[ww * 512 + i];
            sS[i] = s;
        }
        __syncthreads();

        // --- online softmax (wave 0, one lane per query row) ---
        if (w == 0 && lane < 16) {
            const int r = lane;
            float mb = -3.0e38f;
            for (int c = 0; c < 32; ++c) mb = fmaxf(mb, sS[r * 32 + c]);
            const float m_new = fmaxf(m_run, mb);
            const float corr = exp2f((m_run - m_new) * LOG2E);
            float psum = 0.0f;
            for (int c = 0; c < 32; ++c) {
                const float p = exp2f((sS[r * 32 + c] - m_new) * LOG2E);
                sP[r * 32 + c] = (__bf16)p;
                psum += p;
            }
            l_run = l_run * corr + psum;
            m_run = m_new;
            sCorr[r] = corr;
        }
        __syncthreads();

        // --- rescale accumulators, then acc += P @ V ---
        {
            const int gh = lane >> 4;
            float cj[8];
#pragma unroll
            for (int jj = 0; jj < 8; ++jj) cj[jj] = sCorr[jj + 8 * gh];
#pragma unroll
            for (int t = 0; t < 8; ++t)
#pragma unroll
                for (int jj = 0; jj < 8; ++jj) acc[t][jj] *= cj[jj];

            v16bf a = load_a_frag(sP, 32, lane);
#pragma unroll
            for (int t = 0; t < 8; ++t) {
                v16bf b = load_b_frag(sVt + (w * 128 + t * 16) * VS, VS, lane);
                acc[t] = wmma_bf16(a, b, acc[t]);
            }
        }
    }

    if (w == 0 && lane < 16) sL[lane] = l_run;
    __syncthreads();
    {
        const int n = lane & 15, gh = lane >> 4;
        float linv[8];
#pragma unroll
        for (int jj = 0; jj < 8; ++jj) linv[jj] = 1.0f / sL[jj + 8 * gh];
        float* Og = O + qrow0 * 1024;
#pragma unroll
        for (int t = 0; t < 8; ++t) {
            const int col = w * 128 + t * 16 + n;
#pragma unroll
            for (int jj = 0; jj < 8; ++jj) {
                const int r = jj + 8 * gh;
                Og[(long long)r * 1024 + col] = acc[t][jj] * linv[jj];
            }
        }
    }
}

// ---------------------------------------------------------------------------
// Launch: x[2,4096,1024] f32, w_qkv[3072,1024] f32, w_proj[1024,1024] f32,
// b_proj[1024] f32 -> out[2,4096,1024] f32.
// Workspace: Q/K/V bf16 (16 MB each) + f32 context (32 MB) = 80 MB.
// ---------------------------------------------------------------------------
extern "C" void kernel_launch(void* const* d_in, const int* in_sizes, int n_in,
                              void* d_out, int out_size, void* d_ws, size_t ws_size,
                              hipStream_t stream) {
    (void)in_sizes; (void)n_in; (void)out_size; (void)ws_size;
    const float* x      = (const float*)d_in[0];
    const float* w_qkv  = (const float*)d_in[1];
    const float* w_proj = (const float*)d_in[2];
    const float* b_proj = (const float*)d_in[3];
    float* out = (float*)d_out;

    const int Bb = 2, S = 4096, D = 1024;
    const int M = Bb * S;                       // 8192
    const float qscale = 0.03125f;              // 1024^-0.5

    __bf16* qws = (__bf16*)d_ws;
    __bf16* kws = qws + (size_t)M * D;
    __bf16* vws = kws + (size_t)M * D;
    float*  aws = (float*)(vws + (size_t)M * D);

    // 1) QKV projection: [M,1024] x [3072,1024]^T -> bf16 Q(scaled)/K/V
    dim3 g1(3 * D / 128, M / 128);
    gemm_bf16_kernel<<<g1, 256, 0, stream>>>(x, w_qkv, D, 3 * D, 0, qscale,
                                             qws, kws, vws, nullptr, nullptr);

    // 2) Flash attention -> f32 context
    dim3 g2(S / 16, Bb);
    flash_attn_kernel<<<g2, 256, FA_SMEM_BYTES, stream>>>(qws, kws, vws, aws, S);

    // 3) Output projection + bias -> d_out
    dim3 g3(D / 128, M / 128);
    gemm_bf16_kernel<<<g3, 256, 0, stream>>>(aws, w_proj, D, D, 1, 1.0f,
                                             nullptr, nullptr, nullptr, out, b_proj);
}
